// EBTBlock_45586782879846
// MI455X (gfx1250) — compile-verified, matrix operand order checked by reading, not run on hardware
//
#include <hip/hip_runtime.h>
#include <cstddef>

// ---------------------------------------------------------------------------
// DiT adaLN-Zero block for MI455X (gfx1250, wave32, WMMA bf16 16x16x32)
// B=8, S=1024, D=1024, H=16, DH=64  ->  M = B*S = 8192 tokens
// ---------------------------------------------------------------------------

typedef unsigned short u16;
typedef unsigned int   u32;
typedef __attribute__((ext_vector_type(4)))  u32    v4u;
typedef __attribute__((ext_vector_type(4)))  float  v4f;
typedef __attribute__((ext_vector_type(8)))  float  v8f;
typedef __attribute__((ext_vector_type(16))) __bf16 v16bf;

#define CB 8
#define CS 1024
#define CD 1024
#define CH 16
#define CDH 64
#define CM (CB * CS)          // 8192 rows
#define MODSTRIDE (6 * CD)    // 6144 per batch

union FragB {                 // one WMMA bf16 A/B operand (16 bf16 per lane)
    v4u   q[2];
    u16   h[16];
    v16bf bf;
};

__device__ __forceinline__ u16 f2bf(float f) {
    u32 u = __builtin_bit_cast(u32, f);
    u += 0x7fffu + ((u >> 16) & 1u);      // round-to-nearest-even
    return (u16)(u >> 16);
}

__device__ __forceinline__ float gelu_tanh(float x) {
    float c = 0.7978845608028654f * (x + 0.044715f * x * x * x);
    return 0.5f * x * (1.0f + tanhf(c));
}

__device__ __forceinline__ v8f wmma_bf16(v16bf a, v16bf b, v8f c) {
    // D = A(16x32) x B(32x16) + C, f32 accumulate
    return __builtin_amdgcn_wmma_f32_16x16x32_bf16(
        false, a, false, b, (short)0, c, false, false);
}

// ---------------------------------------------------------------------------
// 1) mods[b, j] = SiLU(y[b,:]) . w_adaln[j,:] + b_adaln[j]     (j < 6144)
// ---------------------------------------------------------------------------
__global__ __launch_bounds__(256) void mods_kernel(
    const float* __restrict__ y, const float* __restrict__ w,
    const float* __restrict__ bias, float* __restrict__ mods)
{
    __shared__ float sy[CD];
    int b = blockIdx.y;
    int j = blockIdx.x * 256 + threadIdx.x;
    const float* yr = y + (size_t)b * CD;
    for (int i = threadIdx.x; i < CD; i += 256) {
        float v = yr[i];
        sy[i] = v / (1.0f + __expf(-v));      // SiLU
    }
    __syncthreads();
    const float* wr = w + (size_t)j * CD;
    float acc = bias[j];
    for (int d = 0; d < CD; d += 4) {
        v4f wv = *(const v4f*)(wr + d);
        v4f yv = *(const v4f*)(sy + d);
        acc += wv[0]*yv[0] + wv[1]*yv[1] + wv[2]*yv[2] + wv[3]*yv[3];
    }
    mods[(size_t)b * MODSTRIDE + j] = acc;
}

// ---------------------------------------------------------------------------
// 2) LayerNorm (no affine) + adaLN modulate -> bf16
//    out = LN(x) * (1 + sc[b,:]) + sh[b,:]
// ---------------------------------------------------------------------------
__global__ __launch_bounds__(256) void ln_mod_kernel(
    const float* __restrict__ x, const float* __restrict__ mods,
    u16* __restrict__ out, int sh_chunk, int sc_chunk)
{
    int row = blockIdx.x;           // token row in [0, M)
    int b   = row >> 10;            // S = 1024
    int t   = threadIdx.x;
    const float* xr = x + (size_t)row * CD;
    v4f xv = *(const v4f*)(xr + t * 4);
    float s  = xv[0] + xv[1] + xv[2] + xv[3];
    float ss = xv[0]*xv[0] + xv[1]*xv[1] + xv[2]*xv[2] + xv[3]*xv[3];
    for (int m = 16; m >= 1; m >>= 1) {
        s  += __shfl_xor(s,  m);
        ss += __shfl_xor(ss, m);
    }
    __shared__ float red[18];
    int w = t >> 5, lane = t & 31;
    if (lane == 0) { red[w] = s; red[8 + w] = ss; }
    __syncthreads();
    if (t == 0) {
        float s1 = 0.f, s2 = 0.f;
        for (int i = 0; i < 8; ++i) { s1 += red[i]; s2 += red[8 + i]; }
        float mean = s1 * (1.0f / CD);
        float var  = s2 * (1.0f / CD) - mean * mean;
        red[16] = mean;
        red[17] = rsqrtf(var + 1e-5f);
    }
    __syncthreads();
    float mean = red[16], rstd = red[17];
    const float* shp = mods + (size_t)b * MODSTRIDE + sh_chunk * CD;
    const float* scp = mods + (size_t)b * MODSTRIDE + sc_chunk * CD;
    u16* orow = out + (size_t)row * CD;
    for (int i = 0; i < 4; ++i) {
        int c = t * 4 + i;
        float r = (xv[i] - mean) * rstd * (1.0f + scp[c]) + shp[c];
        orow[c] = f2bf(r);
    }
}

// ---------------------------------------------------------------------------
// 3) WMMA GEMM:  C[M,N] = A_bf16[M,K] . W_f32[N,K]^T  (+ epilogue)
//    Block: 256 thr / 8 waves, tile 128x128, K staged 64-wide in LDS with
//    register-prefetch pipelining (global loads of chunk k+1 overlap the 16
//    WMMAs of chunk k).
//    EPI: 0=store bf16 (qkv)   1=proj: res + gate*(acc+bias) -> f32
//         2=fc1: gelu(acc+bias) -> bf16   3=fc2: res + gate*(acc+bias) -> f32
// ---------------------------------------------------------------------------
template <int EPI>
__global__ __launch_bounds__(256) void gemm_bf16_kernel(
    const u16*  __restrict__ A, const float* __restrict__ W,
    const float* __restrict__ bias, const float* __restrict__ mods,
    const float* __restrict__ resid, void* __restrict__ outv,
    int N, int K, int gate_chunk)
{
    constexpr int AST = 72;                  // LDS row stride (halves), 144B
    __shared__ u16 Ab[128 * AST];
    __shared__ u16 Wb[128 * AST];

    int t = threadIdx.x, lane = t & 31, wave = t >> 5;
    int nBase = blockIdx.x * 128, mBase = blockIdx.y * 128;
    int wm = wave & 3, wn = wave >> 2;       // 4 waves in M x 2 in N
    int m0 = wm * 32, n0 = wn * 64;
    int colL  = lane & 15;
    int khoff = (lane >> 4) << 3;            // 0 or 8

    v8f acc[2][4];
    v8f zero8 = {0.f,0.f,0.f,0.f,0.f,0.f,0.f,0.f};
    for (int mi = 0; mi < 2; ++mi)
        for (int ni = 0; ni < 4; ++ni) acc[mi][ni] = zero8;

    // register-prefetch pipeline state
    int arow = t >> 1, aseg = t & 1;         // A: 128 rows x 2 segs of 32 halves
    int wrow = t >> 2, wseg = t & 3;         // W: 64 rows x 4 segs of 16 f32
    const u16*   aSrc = A + (size_t)(mBase + arow) * K + aseg * 32;
    const float* wSrc = W + (size_t)(nBase + wrow) * K + wseg * 16;
    v4u aR[4];
    v4f wR[4];
    for (int i = 0; i < 4; ++i) aR[i] = *(const v4u*)(aSrc + i * 8);
    for (int p = 0; p < 4; ++p) wR[p] = *(const v4f*)(wSrc + p * 4);

    int nch = K >> 6;
    for (int kc = 0; kc < nch; ++kc) {
        {   // stash prefetched chunk into LDS (W converted f32 -> bf16)
            v4u* ad = (v4u*)&Ab[arow * AST + aseg * 32];
            ad[0] = aR[0]; ad[1] = aR[1]; ad[2] = aR[2]; ad[3] = aR[3];
            union { u16 h[16]; v4u q[2]; } pk;
            for (int p = 0; p < 4; ++p) {
                pk.h[p*4+0] = f2bf(wR[p][0]); pk.h[p*4+1] = f2bf(wR[p][1]);
                pk.h[p*4+2] = f2bf(wR[p][2]); pk.h[p*4+3] = f2bf(wR[p][3]);
            }
            v4u* wd = (v4u*)&Wb[wrow * AST + wseg * 16];
            wd[0] = pk.q[0]; wd[1] = pk.q[1];
        }
        __syncthreads();

        if (kc + 1 < nch) {                  // prefetch next chunk (overlaps WMMA)
            const u16*   as = aSrc + (size_t)(kc + 1) * 64;
            const float* ws = wSrc + (size_t)(kc + 1) * 64;
            for (int i = 0; i < 4; ++i) aR[i] = *(const v4u*)(as + i * 8);
            for (int p = 0; p < 4; ++p) wR[p] = *(const v4f*)(ws + p * 4);
        }

        for (int kk = 0; kk < 2; ++kk) {     // two K=32 WMMA steps
            FragB af[2], wf[4];
            for (int mi = 0; mi < 2; ++mi) {
                int base = (m0 + mi * 16 + colL) * AST + kk * 32 + khoff;
                af[mi].q[0] = *(const v4u*)&Ab[base];
                af[mi].q[1] = *(const v4u*)&Ab[base + 16];
            }
            for (int ni = 0; ni < 4; ++ni) {
                int base = (n0 + ni * 16 + colL) * AST + kk * 32 + khoff;
                wf[ni].q[0] = *(const v4u*)&Wb[base];
                wf[ni].q[1] = *(const v4u*)&Wb[base + 16];
            }
            for (int mi = 0; mi < 2; ++mi)
                for (int ni = 0; ni < 4; ++ni)
                    acc[mi][ni] = wmma_bf16(af[mi].bf, wf[ni].bf, acc[mi][ni]);
        }
        __syncthreads();
    }

    // epilogue: C layout -> lane holds (m = base+v+khoff, n = base+colL)
    for (int mi = 0; mi < 2; ++mi) {
        for (int ni = 0; ni < 4; ++ni) {
            int n = nBase + n0 + ni * 16 + colL;
            float bv = 0.f;
            if constexpr (EPI != 0) bv = bias[n];
            for (int v = 0; v < 8; ++v) {
                int m = mBase + m0 + mi * 16 + v + khoff;
                float val = acc[mi][ni][v];
                if constexpr (EPI == 0) {
                    ((u16*)outv)[(size_t)m * N + n] = f2bf(val);
                } else if constexpr (EPI == 2) {
                    ((u16*)outv)[(size_t)m * N + n] = f2bf(gelu_tanh(val + bv));
                } else {   // 1 or 3: gated residual, N == D == 1024
                    int bb = m >> 10;
                    float g = mods[(size_t)bb * MODSTRIDE + gate_chunk * CD + n];
                    float r = resid[(size_t)m * CD + n];
                    ((float*)outv)[(size_t)m * CD + n] = r + g * (val + bv);
                }
            }
        }
    }
}

// ---------------------------------------------------------------------------
// 4) Flash attention, transposed layout. 4 waves per block: each wave owns 16
//    queries of one (b, h); the 32x64 V block is staged transposed into LDS
//    once per block and shared by all 4 waves. S^T = K.Q^T keeps the softmax
//    stats per-lane; the S^T accumulator registers are reused directly as the
//    P^T B-operand; O^T = V^T.P^T.   qkv: bf16 [B,S,3D]; o: bf16 [B,S,D].
// ---------------------------------------------------------------------------
__global__ __launch_bounds__(128) void attn_kernel(
    const u16* __restrict__ qkv, u16* __restrict__ o)
{
    constexpr int VT = 40;                 // V^T LDS row stride (halves), 80B
    __shared__ u16 Vt[64 * VT];

    int t     = threadIdx.x;
    int lane  = t & 31;
    int wave  = t >> 5;
    int qbase = blockIdx.x * 64 + wave * 16;
    int h     = blockIdx.y;
    int b     = blockIdx.z;
    int colL  = lane & 15;
    int khoff = (lane >> 4) << 3;

    // Q^T B-fragments (64 K-depth = two frags), loaded once per wave
    FragB qf[2];
    {
        const u16* qptr = qkv + (size_t)(b * CS + qbase + colL) * (3 * CD) + h * CDH;
        for (int kh = 0; kh < 2; ++kh) {
            int base = kh * 32 + khoff;
            qf[kh].q[0] = *(const v4u*)(qptr + base);
            qf[kh].q[1] = *(const v4u*)(qptr + base + 16);
        }
    }

    v8f zero8 = {0.f,0.f,0.f,0.f,0.f,0.f,0.f,0.f};
    v8f ot[4] = {zero8, zero8, zero8, zero8};     // O^T: 4 dh tiles x (16x16)
    float mrun = -3.0e38f, lrun = 0.0f;

    // staging roles: thread -> V row j = t>>2, dh quarter = t&3
    int vj = t >> 2, vq = t & 3;

    for (int jb = 0; jb < CS / 32; ++jb) {
        int jbase = jb * 32;

        // stage V^T cooperatively (128 threads, 32x64 block)
        {
            const u16* vptr = qkv + (size_t)(b * CS + jbase + vj) * (3 * CD)
                              + 2 * CD + h * CDH + vq * 16;
            union { v4u q[2]; u16 h[16]; } vb;
            vb.q[0] = *(const v4u*)(vptr);
            vb.q[1] = *(const v4u*)(vptr + 8);
            for (int i = 0; i < 16; ++i) Vt[(vq * 16 + i) * VT + vj] = vb.h[i];
        }
        __syncthreads();

        // S^T tiles: rows j (2 tiles of 16), cols q;  A = K rows, B = Q^T
        v8f st[2] = {zero8, zero8};
        for (int jt = 0; jt < 2; ++jt) {
            const u16* kptr = qkv
                + (size_t)(b * CS + jbase + jt * 16 + colL) * (3 * CD)
                + CD + h * CDH;
            for (int kh = 0; kh < 2; ++kh) {
                FragB kf;
                int base = kh * 32 + khoff;
                kf.q[0] = *(const v4u*)(kptr + base);
                kf.q[1] = *(const v4u*)(kptr + base + 16);
                st[jt] = wmma_bf16(kf.bf, qf[kh].bf, st[jt]);
            }
        }

        // online softmax (per-lane: this lane owns column q = lane&15)
        float mloc = -3.0e38f;
        for (int jt = 0; jt < 2; ++jt)
            for (int v = 0; v < 8; ++v) {
                st[jt][v] *= 0.125f;                 // 1/sqrt(64)
                mloc = fmaxf(mloc, st[jt][v]);
            }
        mloc = fmaxf(mloc, __shfl_xor(mloc, 16));    // other half's j rows
        float mnew  = fmaxf(mrun, mloc);
        float alpha = __expf(mrun - mnew);

        FragB pf;                                    // P^T B-operand from regs
        float ps = 0.0f;
        for (int v = 0; v < 8; ++v) {
            float p0 = __expf(st[0][v] - mnew);
            float p1 = __expf(st[1][v] - mnew);
            ps += p0 + p1;
            pf.bf[v]     = (__bf16)p0;
            pf.bf[8 + v] = (__bf16)p1;
        }
        ps += __shfl_xor(ps, 16);
        lrun = lrun * alpha + ps;
        mrun = mnew;
        for (int nt = 0; nt < 4; ++nt)
            for (int v = 0; v < 8; ++v) ot[nt][v] *= alpha;

        // O^T += V^T . P^T  (A = V^T from LDS, B = P^T from registers)
        for (int nt = 0; nt < 4; ++nt) {
            FragB vf;
            int base = (nt * 16 + colL) * VT + khoff;
            vf.q[0] = *(const v4u*)&Vt[base];
            vf.q[1] = *(const v4u*)&Vt[base + 16];
            ot[nt] = wmma_bf16(vf.bf, pf.bf, ot[nt]);
        }
        __syncthreads();
    }

    // write O (bf16): lane owns column q; rows dh = nt*16 + v + khoff
    float inv = 1.0f / lrun;
    u16* orow = o + (size_t)(b * CS + qbase + colL) * CD + h * CDH;
    for (int nt = 0; nt < 4; ++nt)
        for (int v = 0; v < 8; ++v)
            orow[nt * 16 + v + khoff] = f2bf(ot[nt][v] * inv);
}

// ---------------------------------------------------------------------------
// Launch
// ---------------------------------------------------------------------------
extern "C" void kernel_launch(void* const* d_in, const int* in_sizes, int n_in,
                              void* d_out, int out_size, void* d_ws, size_t ws_size,
                              hipStream_t stream)
{
    const float* x       = (const float*)d_in[0];
    const float* y       = (const float*)d_in[1];
    const float* w_adaln = (const float*)d_in[2];
    const float* b_adaln = (const float*)d_in[3];
    const float* w_qkv   = (const float*)d_in[4];
    const float* w_proj  = (const float*)d_in[5];
    const float* b_proj  = (const float*)d_in[6];
    const float* w_fc1   = (const float*)d_in[7];
    const float* b_fc1   = (const float*)d_in[8];
    const float* w_fc2   = (const float*)d_in[9];
    const float* b_fc2   = (const float*)d_in[10];
    float* out = (float*)d_out;

    // workspace carve-up (all offsets 256B aligned)
    char* base = (char*)d_ws;
    float* modsf = (float*)(base);                         // 8*6144 f32
    u16*   xm    = (u16*)(base + 196608);                  // 8192x1024 bf16
    u16*   qkvb  = (u16*)(base + 16973824);                // 8192x3072 bf16
    u16*   attno = (u16*)(base + 67305472);                // 8192x1024 bf16
    float* x1    = (float*)(base + 84082688);              // 8192x1024 f32
    u16*   hbuf  = (u16*)(base + 117637120);               // 8192x4096 bf16

    // 1) adaLN modulation params
    mods_kernel<<<dim3(24, CB), 256, 0, stream>>>(y, w_adaln, b_adaln, modsf);

    // 2) LN + modulate (attention branch): chunks sh_a=0, sc_a=1
    ln_mod_kernel<<<CM, 256, 0, stream>>>(x, modsf, xm, 0, 1);

    // 3) QKV GEMM: [8192,1024] x [3072,1024]^T -> bf16
    gemm_bf16_kernel<0><<<dim3(3 * CD / 128, CM / 128), 256, 0, stream>>>(
        xm, w_qkv, nullptr, modsf, nullptr, qkvb, 3 * CD, CD, 0);

    // 4) flash attention (4 waves/block, shared V^T staging)
    attn_kernel<<<dim3(CS / 64, CH, CB), 128, 0, stream>>>(qkvb, attno);

    // 5) proj GEMM + gated residual (gate g_a = chunk 2): x1 = x + g_a*(o@Wp^T+b)
    gemm_bf16_kernel<1><<<dim3(CD / 128, CM / 128), 256, 0, stream>>>(
        attno, w_proj, b_proj, modsf, x, x1, CD, CD, 2);

    // 6) LN + modulate (MLP branch): chunks sh_m=3, sc_m=4
    ln_mod_kernel<<<CM, 256, 0, stream>>>(x1, modsf, xm, 3, 4);

    // 7) FC1 GEMM + GELU(tanh) -> bf16
    gemm_bf16_kernel<2><<<dim3(4 * CD / 128, CM / 128), 256, 0, stream>>>(
        xm, w_fc1, b_fc1, modsf, nullptr, hbuf, 4 * CD, CD, 0);

    // 8) FC2 GEMM + gated residual (gate g_m = chunk 5) -> f32 output
    gemm_bf16_kernel<3><<<dim3(CD / 128, CM / 128), 256, 0, stream>>>(
        hbuf, w_fc2, b_fc2, modsf, x1, out, CD, 4 * CD, 5);

    (void)in_sizes; (void)n_in; (void)out_size; (void)ws_size;
}
